// ClusterMemoryAMP_16234976378943
// MI455X (gfx1250) — compile-verified, matrix operand order checked by reading, not run on hardware
//
#include <hip/hip_runtime.h>
#include <hip/hip_bf16.h>

// ---------------------------------------------------------------------------
// ClusterMemory loss on gfx1250 (MI455X)
//   inputs   [1024,256] f32, targets [1024] int, features [100000,256] f32
//   out      scalar f32 loss
// Pipeline (all on `stream`):
//   k_init      : zero sum-exp accumulators + output
//   k_normalize : row-normalize x, store fp32 + bf16 copies
//   k_cvt_feats : features fp32 -> bf16 (51.2 MB, L2-resident afterwards)
//   k_gemm      : register-blocked bf16 WMMA GEMM: each wave owns 32 rows x
//                 128 cols (16 accum tiles); every B fragment feeds 2 WMMAs
//                 -> 1.25 loads/WMMA instead of 2.25; branchless, immediate
//                 offsets; sum(exp(20*logit)) per (row, half) via f32 atomics
//   k_finalize  : exact fp32 target logits + log-sum-exp -> scalar loss
//
// Workspace requirement: ~52.8 MB + 64 KB slack (ragged last N-tile
// intentionally over-reads < 48 KB past the bf16 feature copy; those tiles'
// contributions are discarded in the epilogue).
// ---------------------------------------------------------------------------

typedef __attribute__((ext_vector_type(16))) __bf16 v16bf;
typedef __attribute__((ext_vector_type(4)))  __bf16 v4bf;
typedef __attribute__((ext_vector_type(8)))  float  v8f;

#define TEMP_INV 20.0f      // 1 / 0.05
#define B_ROWS   1024
#define DFEAT    256
#define NCLUST   50000
#define NTOT     100000     // 2 * NCLUST
#define BM       128        // WG rows  (4 row-waves x 32 rows)
#define BN       256        // WG cols  (2 col-waves x 128 cols)
#define NTILES   ((NTOT + BN - 1) / BN)   // 391

// ---------------------------------------------------------------- init ------
__global__ void k_init(float* __restrict__ S, float* __restrict__ out) {
    for (int j = threadIdx.x; j < B_ROWS * 2; j += 256) S[j] = 0.0f;
    if (threadIdx.x == 0) out[0] = 0.0f;
}

// ----------------------------------------------------------- normalize -----
__global__ void __launch_bounds__(256)
k_normalize(const float* __restrict__ in, float* __restrict__ xf,
            __bf16* __restrict__ xb) {
    const int wave = threadIdx.x >> 5;
    const int lane = threadIdx.x & 31;
    const int row  = blockIdx.x * 8 + wave;

    const float* p = in + (size_t)row * DFEAT;
    float v[8];
    float ss = 0.0f;
#pragma unroll
    for (int j = 0; j < 8; ++j) {
        v[j] = p[lane + 32 * j];
        ss += v[j] * v[j];
    }
#pragma unroll
    for (int off = 16; off > 0; off >>= 1) ss += __shfl_xor(ss, off, 32);

    const float rn = rsqrtf(ss);
    float*  q  = xf + (size_t)row * DFEAT;
    __bf16* qb = xb + (size_t)row * DFEAT;
#pragma unroll
    for (int j = 0; j < 8; ++j) {
        const float x = v[j] * rn;
        q[lane + 32 * j]  = x;
        qb[lane + 32 * j] = (__bf16)x;
    }
}

// ------------------------------------------------------- feature convert ---
__global__ void __launch_bounds__(256)
k_cvt_feats(const float* __restrict__ f, __bf16* __restrict__ fb) {
    const size_t i = ((size_t)blockIdx.x * 256 + threadIdx.x) * 4;
    const float4 v = *(const float4*)(f + i);
    v4bf o;
    o.x = (__bf16)v.x; o.y = (__bf16)v.y; o.z = (__bf16)v.z; o.w = (__bf16)v.w;
    *(v4bf*)(fb + i) = o;
}

// ------------------------------------------------------------- WMMA GEMM ---
// grid = (391 N-tiles, 8 M-tiles), block = 256 (8 waves as 4 row x 2 col)
// wave (wr,wc): rows [blockIdx.y*128 + wr*32, +32),
//               cols [blockIdx.x*256 + wc*128, +128)
// Per K-step: 2 A fragments + 8 B fragments; each B fragment is consumed by
// two WMMAs (register reuse) -> 128 WMMAs per wave, 1.25 load-pairs/WMMA.
__global__ void __launch_bounds__(256)
k_gemm(const __bf16* __restrict__ xb, const __bf16* __restrict__ fb,
       float* __restrict__ S) {
    const int lane  = threadIdx.x & 31;
    const int wave  = threadIdx.x >> 5;
    const int wr    = wave & 3;         // row-wave
    const int wc    = wave >> 2;        // col-wave
    const int nbase = blockIdx.x * BN + wc * 128;
    const int row0  = blockIdx.y * BM + wr * 32;
    const int lrow  = lane & 15;        // A: row in tile / B: col in tile
    const int hi    = lane >> 4;        // which K-half this lane carries

    // per-lane fragment bases (bf16 A/B layout: lanes 0-15 carry K {0..7,16..23},
    // lanes 16-31 carry K {8..15,24..31} -> +hi*8 elements)
    const __bf16* abase = xb + (size_t)(row0 + lrow) * DFEAT + hi * 8;
    const __bf16* bbase = fb + (size_t)(nbase + lrow) * DFEAT + hi * 8;

    v8f acc[2][8] = {};                 // [row-tile][col-tile]
    union Frag { v16bf v; uint4 q[2]; };

#pragma unroll
    for (int kt = 0; kt < 8; ++kt) {    // K = 256 in steps of 32
        Frag a0, a1;                    // rows rt*16: +rt*8192 B, +kt*64 B
        a0.q[0] = *(const uint4*)(abase + kt * 32);
        a0.q[1] = *(const uint4*)(abase + kt * 32 + 16);
        a1.q[0] = *(const uint4*)(abase + 16 * DFEAT + kt * 32);
        a1.q[1] = *(const uint4*)(abase + 16 * DFEAT + kt * 32 + 16);
#pragma unroll
        for (int ct = 0; ct < 8; ++ct) {   // offsets: ct*8192 + kt*64 B
            Frag b;
            b.q[0] = *(const uint4*)(bbase + ct * 16 * DFEAT + kt * 32);
            b.q[1] = *(const uint4*)(bbase + ct * 16 * DFEAT + kt * 32 + 16);
            acc[0][ct] = __builtin_amdgcn_wmma_f32_16x16x32_bf16(
                false, a0.v, false, b.v, (short)0, acc[0][ct], false, false);
            acc[1][ct] = __builtin_amdgcn_wmma_f32_16x16x32_bf16(
                false, a1.v, false, b.v, (short)0, acc[1][ct], false, false);
        }
    }

    // ---- epilogue: sum(exp(20*logit)) per row, per half (mean | hard) ----
#pragma unroll
    for (int rt = 0; rt < 2; ++rt) {
        float s0[8], s1[8];
#pragma unroll
        for (int v = 0; v < 8; ++v) { s0[v] = 0.0f; s1[v] = 0.0f; }

#pragma unroll
        for (int ct = 0; ct < 8; ++ct) {
            const int ncb = nbase + ct * 16;
            if (ncb >= NTOT) continue;           // wave-uniform (ragged WG)
            const bool hhalf = (ncb >= NCLUST);  // 50000 % 16 == 0 -> uniform
#pragma unroll
            for (int v = 0; v < 8; ++v) {
                const float e = __expf(acc[rt][ct][v] * TEMP_INV);
                if (hhalf) s1[v] += e; else s0[v] += e;
            }
        }

        // C layout: lanes 0-15 hold rows v (VGPR v), lanes 16-31 rows v+8
#pragma unroll
        for (int v = 0; v < 8; ++v) {
#pragma unroll
            for (int off = 1; off < 16; off <<= 1) {
                s0[v] += __shfl_xor(s0[v], off, 16);
                s1[v] += __shfl_xor(s1[v], off, 16);
            }
        }
        if (lrow == 0) {
            const int rbase = row0 + rt * 16 + (hi ? 8 : 0);
#pragma unroll
            for (int v = 0; v < 8; ++v) {
                if (s0[v] != 0.0f) atomicAdd(&S[(rbase + v) * 2 + 0], s0[v]);
                if (s1[v] != 0.0f) atomicAdd(&S[(rbase + v) * 2 + 1], s1[v]);
            }
        }
    }
}

// -------------------------------------------------------------- finalize ---
__global__ void __launch_bounds__(256)
k_finalize(const float* __restrict__ xf, const float* __restrict__ feats,
           const int* __restrict__ tgt, const float* __restrict__ S,
           float* __restrict__ out) {
    const int wave = threadIdx.x >> 5;
    const int lane = threadIdx.x & 31;
    const int row  = blockIdx.x * 8 + wave;
    const int t    = tgt[row];

    const float* xp = xf + (size_t)row * DFEAT;
    const float* fm = feats + (size_t)t * DFEAT;            // mean half
    const float* fh = feats + (size_t)(NCLUST + t) * DFEAT; // hard half

    float d0 = 0.0f, d1 = 0.0f;
#pragma unroll
    for (int j = 0; j < 8; ++j) {
        const float x = xp[lane + 32 * j];
        d0 += x * fm[lane + 32 * j];
        d1 += x * fh[lane + 32 * j];
    }
#pragma unroll
    for (int off = 16; off > 0; off >>= 1) {
        d0 += __shfl_xor(d0, off, 32);
        d1 += __shfl_xor(d1, off, 32);
    }
    if (lane == 0) {
        const float loss =
            0.5f * ((logf(S[row * 2 + 0]) - d0 * TEMP_INV) +
                    (logf(S[row * 2 + 1]) - d1 * TEMP_INV));
        atomicAdd(out, loss * (1.0f / (float)B_ROWS));
    }
}

// ---------------------------------------------------------------------------
extern "C" void kernel_launch(void* const* d_in, const int* in_sizes, int n_in,
                              void* d_out, int out_size, void* d_ws, size_t ws_size,
                              hipStream_t stream) {
    const float* d_inputs  = (const float*)d_in[0];
    const int*   d_targets = (const int*)d_in[1];
    const float* d_feats   = (const float*)d_in[2];
    float*       d_loss    = (float*)d_out;

    // workspace layout (all 16B aligned); fb needs ~64KB slack after it
    char* ws = (char*)d_ws;
    float*  S  = (float*)ws;                                  //   8 KB
    float*  xf = (float*)(ws + 8192);                         //   1 MB
    __bf16* xb = (__bf16*)(ws + 8192 + 1048576);              // 512 KB
    __bf16* fb = (__bf16*)(ws + 8192 + 1048576 + 524288);     // 51.2 MB

    k_init<<<1, 256, 0, stream>>>(S, d_loss);
    k_normalize<<<B_ROWS / 8, 256, 0, stream>>>(d_inputs, xf, xb);
    k_cvt_feats<<<(NTOT * DFEAT) / (256 * 4), 256, 0, stream>>>(d_feats, fb);
    k_gemm<<<dim3(NTILES, B_ROWS / BM), 256, 0, stream>>>(xb, fb, S);
    k_finalize<<<B_ROWS / 8, 256, 0, stream>>>(xf, d_feats, d_targets, S, d_loss);
}
